// VLAAdaptiveTransformer_68685116997682
// MI455X (gfx1250) — matrix operand order repaired and checked by reading, not compiled, not run on hardware
//
#include <hip/hip_runtime.h>
#include <math.h>

// ---------------- model constants ----------------
#define B_   4
#define SV_  256
#define SL_  256
#define S_   512
#define D_   1024
#define H_   16
#define HD_  64
#define F_   4096
#define CH_  16
#define SC_  528           // CH_ + S_
#define ADIM_ 7

typedef __bf16 bf16;
typedef bf16  bf16x16 __attribute__((ext_vector_type(16)));
typedef float f32x8   __attribute__((ext_vector_type(8)));
typedef unsigned int u32x4 __attribute__((ext_vector_type(4)));

union FragU { u32x4 q[2]; bf16x16 v; };

__device__ __forceinline__ float gelu_f(float x) {
    return 0.5f * x * (1.0f + erff(x * 0.70710678118654752f));
}

// ================= WMMA GEMM: C(M,N) = A(M,K)bf16 @ Bt(N,K)bf16^T =================
// EPI: 0 = none, 1 = +bias, 2 = +bias,GELU, 3 = +bias,+residual
#define BM 128
#define BN 128
#define BK 32

template <int EPI>
__global__ __launch_bounds__(256)
void gemm_bt_kernel(const bf16* __restrict__ A,  long lda, long a1, long a2,
                    const bf16* __restrict__ Bt, long ldb, long b1, long b2,
                    float* __restrict__ C, long ldc, long c1, long c2,
                    bf16* __restrict__ Cb,
                    const float* __restrict__ bias,
                    const float* __restrict__ Res, long ldr,
                    int M, int N, int K, int zdiv)
{
    __shared__ alignas(16) bf16 As[BM * BK];
    __shared__ alignas(16) bf16 Bs[BN * BK];

    const int tid  = threadIdx.x;
    const int z    = blockIdx.z;
    const int zq   = z / zdiv, zr = z % zdiv;
    A  += (long)zq * a1 + (long)zr * a2;
    Bt += (long)zq * b1 + (long)zr * b2;
    const long coff = (long)zq * c1 + (long)zr * c2;

    const int m0 = blockIdx.y * BM;
    const int n0 = blockIdx.x * BN;

    const int wave = tid >> 5;
    const int lane = tid & 31;
    const int wm   = wave & 1;   // 2 waves along M (64 rows each)
    const int wn   = wave >> 1;  // 4 waves along N (32 cols each)
    const int half = lane >> 4;  // lane half selects K sub-range per ISA layout
    const int l16  = lane & 15;

    f32x8 acc[4][2] = {};

    const int lr = tid >> 2;        // 0..63
    const int lc = (tid & 3) * 8;   // 0,8,16,24 (bf16 elements)

    for (int k0 = 0; k0 < K; k0 += BK) {
        __syncthreads();
        // ---- stage A tile (BM x BK) ----
        #pragma unroll
        for (int rr = 0; rr < BM; rr += 64) {
            int row = lr + rr;
            int gm  = m0 + row;
            u32x4 val = {0u, 0u, 0u, 0u};
            if (gm < M && (k0 + lc) < K)
                val = *(const u32x4*)(A + (long)gm * lda + k0 + lc);
            *(u32x4*)(As + row * BK + lc) = val;
        }
        // ---- stage B tile (BN x BK) from (N,K)-major Bt ----
        #pragma unroll
        for (int rr = 0; rr < BN; rr += 64) {
            int row = lr + rr;
            int gn  = n0 + row;
            u32x4 val = {0u, 0u, 0u, 0u};
            if (gn < N && (k0 + lc) < K)
                val = *(const u32x4*)(Bt + (long)gn * ldb + k0 + lc);
            *(u32x4*)(Bs + row * BK + lc) = val;
        }
        // prefetch next K tiles into cache (global_prefetch_b8)
        if (k0 + BK < K) {
            int pa = min(m0 + lr, M - 1);
            int pb = min(n0 + lr, N - 1);
            __builtin_prefetch(A  + (long)pa * lda + k0 + BK, 0, 1);
            __builtin_prefetch(Bt + (long)pb * ldb + k0 + BK, 0, 1);
        }
        __syncthreads();

        // ---- fragments: A 16x32 per ISA (lanes<16: K0-7,16-23; lanes>=16: K8-15,24-31) ----
        bf16x16 af[4], bfr[2];
        #pragma unroll
        for (int f = 0; f < 4; ++f) {
            int row = wm * 64 + f * 16 + l16;
            const bf16* p = As + row * BK;
            FragU u;
            u.q[0] = *(const u32x4*)(p + half * 8);
            u.q[1] = *(const u32x4*)(p + 16 + half * 8);
            af[f] = u.v;
        }
        #pragma unroll
        for (int g = 0; g < 2; ++g) {
            int row = wn * 32 + g * 16 + l16;
            const bf16* p = Bs + row * BK;
            FragU u;
            u.q[0] = *(const u32x4*)(p + half * 8);
            u.q[1] = *(const u32x4*)(p + 16 + half * 8);
            bfr[g] = u.v;
        }
        #pragma unroll
        for (int f = 0; f < 4; ++f)
            #pragma unroll
            for (int g = 0; g < 2; ++g)
                acc[f][g] = __builtin_amdgcn_wmma_f32_16x16x32_bf16(
                    false, af[f], false, bfr[g], (short)0, acc[f][g], false, false);
    }

    // ---- epilogue: C/D 16x16 layout: VGPR r, lanes<16 -> M=r, lanes>=16 -> M=r+8 ----
    #pragma unroll
    for (int g = 0; g < 2; ++g) {
        int n = n0 + wn * 32 + g * 16 + l16;
        if (n >= N) continue;
        float bv = (EPI >= 1) ? bias[n] : 0.0f;
        #pragma unroll
        for (int f = 0; f < 4; ++f) {
            #pragma unroll
            for (int r = 0; r < 8; ++r) {
                int m = m0 + wm * 64 + f * 16 + r + half * 8;
                if (m >= M) continue;
                float v = acc[f][g][r] + bv;
                if (EPI == 2) v = gelu_f(v);
                if (EPI == 3) v += Res[(long)m * ldr + n];
                long idx = coff + (long)m * ldc + n;
                if (C)  C[idx]  = v;
                if (Cb) Cb[idx] = (bf16)v;
            }
        }
    }
}

// ================= LayerNorm: fp32 rows (len 1024) -> fp32 and/or bf16 =================
__global__ __launch_bounds__(256)
void ln_kernel(const float* __restrict__ X, const float* __restrict__ g,
               const float* __restrict__ b, float* __restrict__ Yf,
               bf16* __restrict__ Yb)
{
    __shared__ float r1[8], r2[8];
    long row = blockIdx.x;
    const float* x = X + row * D_;
    int tid = threadIdx.x;

    float s = 0.f;
    for (int i = tid; i < D_; i += 256) s += x[i];
    #pragma unroll
    for (int o = 16; o; o >>= 1) s += __shfl_xor(s, o, 32);
    if ((tid & 31) == 0) r1[tid >> 5] = s;
    __syncthreads();
    float mean = (r1[0]+r1[1]+r1[2]+r1[3]+r1[4]+r1[5]+r1[6]+r1[7]) * (1.0f / D_);

    float v = 0.f;
    for (int i = tid; i < D_; i += 256) { float d = x[i] - mean; v += d * d; }
    #pragma unroll
    for (int o = 16; o; o >>= 1) v += __shfl_xor(v, o, 32);
    if ((tid & 31) == 0) r2[tid >> 5] = v;
    __syncthreads();
    float var  = (r2[0]+r2[1]+r2[2]+r2[3]+r2[4]+r2[5]+r2[6]+r2[7]) * (1.0f / D_);
    float rstd = rsqrtf(var + 1e-5f);

    for (int i = tid; i < D_; i += 256) {
        float y = (x[i] - mean) * rstd * g[i] + b[i];
        if (Yf) Yf[row * D_ + i] = y;
        if (Yb) Yb[row * D_ + i] = (bf16)y;
    }
}

// ================= masked softmax: fp32 scores row -> bf16 probs =================
__global__ __launch_bounds__(256)
void softmax_kernel(const float* __restrict__ Sc, bf16* __restrict__ P,
                    int Skv, int Sq, float scale, int vla_mask)
{
    __shared__ float red[8];
    long row = blockIdx.x;
    int  qi  = (int)(row % (long)Sq);
    const float* s = Sc + row * (long)Skv;
    bf16* p = P + row * (long)Skv;
    int tid = threadIdx.x;

    float mx = -3e38f;
    for (int k = tid; k < Skv; k += 256) {
        float v = s[k] * scale;
        if (vla_mask && !((k <= qi) || (qi < SV_ && k < SV_))) v = -1e9f;
        mx = fmaxf(mx, v);
    }
    #pragma unroll
    for (int o = 16; o; o >>= 1) mx = fmaxf(mx, __shfl_xor(mx, o, 32));
    if ((tid & 31) == 0) red[tid >> 5] = mx;
    __syncthreads();
    mx = fmaxf(fmaxf(fmaxf(red[0], red[1]), fmaxf(red[2], red[3])),
               fmaxf(fmaxf(red[4], red[5]), fmaxf(red[6], red[7])));
    __syncthreads();

    float sum = 0.f;
    for (int k = tid; k < Skv; k += 256) {
        float v = s[k] * scale;
        if (vla_mask && !((k <= qi) || (qi < SV_ && k < SV_))) v = -1e9f;
        sum += __expf(v - mx);
    }
    #pragma unroll
    for (int o = 16; o; o >>= 1) sum += __shfl_xor(sum, o, 32);
    if ((tid & 31) == 0) red[tid >> 5] = sum;
    __syncthreads();
    float tot = red[0]+red[1]+red[2]+red[3]+red[4]+red[5]+red[6]+red[7];
    float inv = 1.0f / tot;

    for (int k = tid; k < Skv; k += 256) {
        float v = s[k] * scale;
        if (vla_mask && !((k <= qi) || (qi < SV_ && k < SV_))) v = -1e9f;
        p[k] = (bf16)(__expf(v - mx) * inv);
    }
}

// ================= transpose+convert: out(N,K)bf16 = in(K,N)fp32^T =================
__global__ __launch_bounds__(256)
void transpose_cvt_kernel(const float* __restrict__ in, long ldin,
                          long i1, long i2, int zdiv,
                          bf16* __restrict__ out, long ldout, long obs,
                          int K, int N)
{
    __shared__ float tile[32][33];
    int z = blockIdx.z;
    in  += (long)(z / zdiv) * i1 + (long)(z % zdiv) * i2;
    out += (long)z * obs;
    int k0 = blockIdx.y * 32, n0 = blockIdx.x * 32;
    int tx = threadIdx.x & 31, ty = threadIdx.x >> 5;
    for (int r = ty; r < 32; r += 8) {
        int k = k0 + r, n = n0 + tx;
        tile[r][tx] = (k < K && n < N) ? in[(long)k * ldin + n] : 0.0f;
    }
    __syncthreads();
    for (int r = ty; r < 32; r += 8) {
        int n = n0 + r, k = k0 + tx;
        if (n < N && k < K) out[(long)n * ldout + k] = (bf16)tile[tx][r];
    }
}

// ================= small elementwise kernels =================
__global__ void cvt_kernel(const float* __restrict__ in, bf16* __restrict__ out, long n)
{
    long i = (long)blockIdx.x * 256 + threadIdx.x;
    if (i < n) out[i] = (bf16)in[i];
}

__global__ __launch_bounds__(256)
void embed_kernel(const int* __restrict__ ids, const int* __restrict__ mids,
                  const float* __restrict__ tok, const float* __restrict__ pos,
                  const float* __restrict__ mod, float* __restrict__ x)
{
    long t = blockIdx.x;                  // b*S + s
    int b = (int)(t / S_), s2 = (int)(t % S_);
    float* xr = x + t * D_;
    int mid = mids[t];
    if (s2 >= SV_) {
        int id = ids[b * SL_ + (s2 - SV_)];
        for (int i = threadIdx.x; i < D_; i += 256)
            xr[i] = tok[(long)id * D_ + i] + pos[(long)(s2 - SV_) * D_ + i] + mod[(long)mid * D_ + i];
    } else {
        for (int i = threadIdx.x; i < D_; i += 256)
            xr[i] += mod[(long)mid * D_ + i];
    }
}

__global__ __launch_bounds__(256)
void gate_kernel(const float* __restrict__ x0, const float* __restrict__ x1,
                 const float* __restrict__ x2, int Kk,
                 const float* __restrict__ gw, const float* __restrict__ gbp, int dp,
                 float* __restrict__ h)
{
    __shared__ float red[8];
    __shared__ float wsh[4];
    long t = blockIdx.x;
    const float* xs[3]; xs[0] = x0; xs[1] = x1; xs[2] = x2;
    int tid = threadIdx.x;
    float sc[3];
    for (int k = 0; k < Kk; ++k) {
        const float* xp = xs[k] + t * D_;
        float d = 0.f;
        for (int i = tid; i < D_; i += 256) d += xp[i] * gw[i];
        #pragma unroll
        for (int o = 16; o; o >>= 1) d += __shfl_xor(d, o, 32);
        if ((tid & 31) == 0) red[tid >> 5] = d;
        __syncthreads();
        float tot = red[0]+red[1]+red[2]+red[3]+red[4]+red[5]+red[6]+red[7];
        sc[k] = tot * (1.0f / 32.0f) + gbp[dp];   // 1/sqrt(1024)
        __syncthreads();
    }
    if (tid == 0) {
        float mx = sc[0];
        for (int k = 1; k < Kk; ++k) mx = fmaxf(mx, sc[k]);
        float s = 0.f;
        for (int k = 0; k < Kk; ++k) { wsh[k] = __expf(sc[k] - mx); s += wsh[k]; }
        for (int k = 0; k < Kk; ++k) wsh[k] /= s;
    }
    __syncthreads();
    for (int i = tid; i < D_; i += 256) {
        float a = 0.f;
        for (int k = 0; k < Kk; ++k) a += wsh[k] * xs[k][t * D_ + i];
        h[t * D_ + i] = a;
    }
}

__global__ __launch_bounds__(256)
void copy_rows_kernel(const float* __restrict__ src, int srcRowsPerB,
                      float* __restrict__ dst, int dstRowsPerB, int dstOff)
{
    long blk = blockIdx.x;
    long b = blk / srcRowsPerB, r = blk % srcRowsPerB;
    const float* s = src + (b * srcRowsPerB + r) * (long)D_;
    float* d = dst + (b * (long)dstRowsPerB + dstOff + r) * (long)D_;
    for (int i = threadIdx.x; i < D_; i += 256) d[i] = s[i];
}

__global__ __launch_bounds__(256)
void slice_add_kernel(float* __restrict__ q, const float* __restrict__ cross)
{
    long blk = blockIdx.x;          // b*CH + r
    long b = blk / CH_, r = blk % CH_;
    float* d = q + blk * (long)D_;
    const float* s = cross + (b * (long)SC_ + r) * (long)D_;
    for (int i = threadIdx.x; i < D_; i += 256) d[i] += s[i];
}

__global__ __launch_bounds__(256)
void bcast_aq_kernel(const float* __restrict__ aq, float* __restrict__ q)
{
    long blk = blockIdx.x;          // b*CH + r
    int r = (int)(blk % CH_);
    for (int i = threadIdx.x; i < D_; i += 256)
        q[blk * (long)D_ + i] = aq[(long)r * D_ + i];
}

// ================= host helpers =================
static void launch_gemm(int epi,
                        const bf16* A, long lda, long a1, long a2,
                        const bf16* Bt, long ldb, long b1, long b2,
                        float* C, long ldc, long c1, long c2, bf16* Cb,
                        const float* bias, const float* Res, long ldr,
                        int M, int N, int K, int Z, int zdiv, hipStream_t st)
{
    dim3 grid((N + BN - 1) / BN, (M + BM - 1) / BM, Z), blk(256);
    switch (epi) {
    case 0: gemm_bt_kernel<0><<<grid, blk, 0, st>>>(A, lda, a1, a2, Bt, ldb, b1, b2, C, ldc, c1, c2, Cb, bias, Res, ldr, M, N, K, zdiv); break;
    case 1: gemm_bt_kernel<1><<<grid, blk, 0, st>>>(A, lda, a1, a2, Bt, ldb, b1, b2, C, ldc, c1, c2, Cb, bias, Res, ldr, M, N, K, zdiv); break;
    case 2: gemm_bt_kernel<2><<<grid, blk, 0, st>>>(A, lda, a1, a2, Bt, ldb, b1, b2, C, ldc, c1, c2, Cb, bias, Res, ldr, M, N, K, zdiv); break;
    default: gemm_bt_kernel<3><<<grid, blk, 0, st>>>(A, lda, a1, a2, Bt, ldb, b1, b2, C, ldc, c1, c2, Cb, bias, Res, ldr, M, N, K, zdiv); break;
    }
}

static void launch_tc(const float* in, long ldin, long i1, long i2, int zdiv,
                      bf16* out, long ldout, long obs, int K, int N, int Z, hipStream_t st)
{
    dim3 g((N + 31) / 32, (K + 31) / 32, Z), b(256);
    transpose_cvt_kernel<<<g, b, 0, st>>>(in, ldin, i1, i2, zdiv, out, ldout, obs, K, N);
}

extern "C" void kernel_launch(void* const* d_in, const int* in_sizes, int n_in,
                              void* d_out, int out_size, void* d_ws, size_t ws_size,
                              hipStream_t stream)
{
    (void)in_sizes; (void)n_in; (void)out_size; (void)ws_size;
    const int*   ids  = (const int*)d_in[0];
    const float* vis  = (const float*)d_in[1];
    const int*   mids = (const int*)d_in[2];
    const float* tok  = (const float*)d_in[3];
    const float* pos  = (const float*)d_in[4];
    const float* mod  = (const float*)d_in[5];
    const float* vp_w1 = (const float*)d_in[6];  const float* vp_b1 = (const float*)d_in[7];
    const float* vp_w2 = (const float*)d_in[8];  const float* vp_b2 = (const float*)d_in[9];
    const float* ln1w = (const float*)d_in[10];  const float* ln1b = (const float*)d_in[11];
    const float* qkvw = (const float*)d_in[12];  const float* qkvbb = (const float*)d_in[13];
    const float* ow   = (const float*)d_in[14];  const float* obb   = (const float*)d_in[15];
    const float* ln2w = (const float*)d_in[16];  const float* ln2b = (const float*)d_in[17];
    const float* f1w  = (const float*)d_in[18];  const float* f1b  = (const float*)d_in[19];
    const float* f2w  = (const float*)d_in[20];  const float* f2b  = (const float*)d_in[21];
    const float* gw   = (const float*)d_in[22];  const float* gb   = (const float*)d_in[23];
    const float* onw  = (const float*)d_in[24];  const float* onb  = (const float*)d_in[25];
    const float* aq   = (const float*)d_in[26];
    const float* snw  = (const float*)d_in[27];  const float* snb  = (const float*)d_in[28];
    const float* saqw = (const float*)d_in[29];  const float* saqb = (const float*)d_in[30];
    const float* saow = (const float*)d_in[31];  const float* saob = (const float*)d_in[32];
    const float* cnw  = (const float*)d_in[33];  const float* cnb  = (const float*)d_in[34];
    const float* caqw = (const float*)d_in[35];  const float* caqb = (const float*)d_in[36];
    const float* caow = (const float*)d_in[37];  const float* caob = (const float*)d_in[38];
    const float* fnw  = (const float*)d_in[39];  const float* fnb  = (const float*)d_in[40];
    const float* a1w  = (const float*)d_in[41];  const float* a1b  = (const float*)d_in[42];
    const float* a2w  = (const float*)d_in[43];  const float* a2b  = (const float*)d_in[44];
    const float* apw  = (const float*)d_in[45];  const float* apb  = (const float*)d_in[46];

    // ---- workspace bump allocator ----
    char* base = (char*)d_ws;
    size_t off = 0;
    auto alloc = [&](size_t bytes) -> void* {
        void* p = base + off;
        off = (off + bytes + 255) & ~(size_t)255;
        return p;
    };
    // bf16 weights, (N,K)-major
    bf16* wqkvT  = (bf16*)alloc(9L * 3072 * 1024 * 2);
    bf16* woT    = (bf16*)alloc(9L * 1024 * 1024 * 2);
    bf16* wf1T   = (bf16*)alloc(9L * 4096 * 1024 * 2);
    bf16* wf2T   = (bf16*)alloc(9L * 1024 * 4096 * 2);
    bf16* vp1T   = (bf16*)alloc(1024L * 1024 * 2);
    bf16* vp2T   = (bf16*)alloc(1024L * 1024 * 2);
    bf16* saqT   = (bf16*)alloc(3072L * 1024 * 2);
    bf16* saoT   = (bf16*)alloc(1024L * 1024 * 2);
    bf16* caqT   = (bf16*)alloc(3072L * 1024 * 2);
    bf16* caoT   = (bf16*)alloc(1024L * 1024 * 2);
    bf16* af1T   = (bf16*)alloc(4096L * 1024 * 2);
    bf16* af2T   = (bf16*)alloc(1024L * 4096 * 2);
    bf16* apT    = (bf16*)alloc((long)ADIM_ * 1024 * 2);
    // activations
    float* x0 = (float*)alloc((long)B_ * S_ * D_ * 4);
    float* x1 = (float*)alloc((long)B_ * S_ * D_ * 4);
    float* x2 = (float*)alloc((long)B_ * S_ * D_ * 4);
    float* x3 = (float*)alloc((long)B_ * S_ * D_ * 4);
    float* xs[4] = {x0, x1, x2, x3};
    float* h      = (float*)alloc((long)B_ * S_ * D_ * 4);
    bf16*  hb     = (bf16*)alloc((long)B_ * SC_ * D_ * 2);
    float* qkvf   = (float*)alloc((long)B_ * SC_ * 3 * D_ * 4);
    bf16*  qkvb   = (bf16*)alloc((long)B_ * SC_ * 3 * D_ * 2);
    float* scoref = (float*)alloc((long)B_ * H_ * SC_ * SC_ * 4);
    bf16*  Pb     = (bf16*)alloc((long)B_ * H_ * SC_ * SC_ * 2);
    bf16*  Vt     = (bf16*)alloc((long)B_ * H_ * HD_ * SC_ * 2);
    float* attnf  = (float*)alloc((long)B_ * SC_ * D_ * 4);
    bf16*  attnb  = (bf16*)alloc((long)B_ * SC_ * D_ * 2);
    bf16*  ffb    = (bf16*)alloc((long)B_ * S_ * F_ * 2);
    bf16*  vfb    = (bf16*)alloc((long)B_ * SV_ * D_ * 2);
    bf16*  t1b    = (bf16*)alloc((long)B_ * SV_ * D_ * 2);
    float* bbf    = (float*)alloc((long)B_ * S_ * D_ * 4);
    float* comb   = (float*)alloc((long)B_ * SC_ * D_ * 4);
    float* crossf = (float*)alloc((long)B_ * SC_ * D_ * 4);
    float* qbuf   = (float*)alloc((long)B_ * CH_ * D_ * 4);
    bf16*  smallb = (bf16*)alloc((long)B_ * CH_ * D_ * 2);

    const int M2048 = B_ * S_;

    // ---- weight convert+transpose to bf16 (N,K) ----
    launch_tc(vp_w1, 1024, 0, 0, 1, vp1T, 1024, 0, 1024, 1024, 1, stream);
    launch_tc(vp_w2, 1024, 0, 0, 1, vp2T, 1024, 0, 1024, 1024, 1, stream);
    launch_tc(qkvw, 3072, 3072L * 1024, 0, 1, wqkvT, 1024, 3072L * 1024, 1024, 3072, 9, stream);
    launch_tc(ow,   1024, 1024L * 1024, 0, 1, woT,   1024, 1024L * 1024, 1024, 1024, 9, stream);
    launch_tc(f1w,  4096, 4096L * 1024, 0, 1, wf1T,  1024, 4096L * 1024, 1024, 4096, 9, stream);
    launch_tc(f2w,  1024, 1024L * 4096, 0, 1, wf2T,  4096, 4096L * 1024, 4096, 1024, 9, stream);
    launch_tc(saqw, 3072, 0, 0, 1, saqT, 1024, 0, 1024, 3072, 1, stream);
    launch_tc(saow, 1024, 0, 0, 1, saoT, 1024, 0, 1024, 1024, 1, stream);
    launch_tc(caqw, 3072, 0, 0, 1, caqT, 1024, 0, 1024, 3072, 1, stream);
    launch_tc(caow, 1024, 0, 0, 1, caoT, 1024, 0, 1024, 1024, 1, stream);
    launch_tc(a1w,  4096, 0, 0, 1, af1T, 1024, 0, 1024, 4096, 1, stream);
    launch_tc(a2w,  1024, 0, 0, 1, af2T, 4096, 0, 4096, 1024, 1, stream);
    launch_tc(apw,  ADIM_, 0, 0, 1, apT, 1024, 0, 1024, ADIM_, 1, stream);

    // ---- vision projector -> x0[:, :SV, :] ----
    long nvf = (long)B_ * SV_ * D_;
    cvt_kernel<<<(int)((nvf + 255) / 256), 256, 0, stream>>>(vis, vfb, nvf);
    launch_gemm(2, vfb, 1024, 0, 0, vp1T, 1024, 0, 0,
                nullptr, 1024, 0, 0, t1b, vp_b1, nullptr, 0,
                B_ * SV_, 1024, 1024, 1, 1, stream);
    launch_gemm(1, t1b, 1024, (long)SV_ * 1024, 0, vp2T, 1024, 0, 0,
                x0, 1024, (long)S_ * D_, 0, nullptr, vp_b2, nullptr, 0,
                SV_, 1024, 1024, B_, 1, stream);
    // ---- language embed + modality add (all rows) ----
    embed_kernel<<<B_ * S_, 256, 0, stream>>>(ids, mids, tok, pos, mod, x0);

    // ---- 3 blocks x 3 layers ----
    for (int bi = 0; bi < 3; ++bi) {
        int dp = bi + 1;
        gate_kernel<<<B_ * S_, 256, 0, stream>>>(
            xs[0], bi >= 1 ? xs[1] : xs[0], bi >= 2 ? xs[2] : xs[0],
            bi + 1, gw + (long)dp * D_, gb, dp, h);
        for (int li = 0; li < 3; ++li) {
            int i = bi * 3 + li;
            // MHA
            ln_kernel<<<M2048, 256, 0, stream>>>(h, ln1w + (long)i * D_, ln1b + (long)i * D_, nullptr, hb);
            launch_gemm(1, hb, 1024, 0, 0, wqkvT + (long)i * 3072 * 1024, 1024, 0, 0,
                        qkvf, 3072, 0, 0, qkvb, qkvbb + (long)i * 3072, nullptr, 0,
                        M2048, 3072, 1024, 1, 1, stream);
            launch_gemm(0, qkvb, 3072, (long)S_ * 3072, 64,
                        qkvb + 1024, 3072, (long)S_ * 3072, 64,
                        scoref, S_, (long)H_ * S_ * S_, (long)S_ * S_, nullptr,
                        nullptr, nullptr, 0, S_, S_, HD_, B_ * H_, H_, stream);
            softmax_kernel<<<B_ * H_ * S_, 256, 0, stream>>>(scoref, Pb, S_, S_, 0.125f, 1);
            launch_tc(qkvf + 2048, 3072, (long)S_ * 3072, 64, H_,
                      Vt, S_, (long)HD_ * S_, S_, HD_, B_ * H_, stream);
            launch_gemm(0, Pb, S_, (long)H_ * S_ * S_, (long)S_ * S_,
                        Vt, S_, (long)H_ * HD_ * S_, (long)HD_ * S_,
                        nullptr, 1024, (long)S_ * D_, 64, attnb,
                        nullptr, nullptr, 0, S_, HD_, S_, B_ * H_, H_, stream);
            launch_gemm(3, attnb, 1024, 0, 0, woT + (long)i * 1024 * 1024, 1024, 0, 0,
                        h, 1024, 0, 0, nullptr, obb + (long)i * D_, h, 1024,
                        M2048, 1024, 1024, 1, 1, stream);
            // FFN
            ln_kernel<<<M2048, 256, 0, stream>>>(h, ln2w + (long)i * D_, ln2b + (long)i * D_, nullptr, hb);
            launch_gemm(2, hb, 1024, 0, 0, wf1T + (long)i * 4096 * 1024, 1024, 0, 0,
                        nullptr, 4096, 0, 0, ffb, f1b + (long)i * F_, nullptr, 0,
                        M2048, 4096, 1024, 1, 1, stream);
            float* outp = (li == 2) ? xs[bi + 1] : h;
            launch_gemm(3, ffb, 4096, 0, 0, wf2T + (long)i * 1024 * 4096, 4096, 0, 0,
                        outp, 1024, 0, 0, nullptr, f2b + (long)i * D_, h, 1024,
                        M2048, 1024, 4096, 1, 1, stream);
        }
    }

    // ---- backbone norm ----
    ln_kernel<<<M2048, 256, 0, stream>>>(x3, onw, onb, bbf, nullptr);

    // ---- action head: self-attention over CHUNK queries ----
    bcast_aq_kernel<<<B_ * CH_, 256, 0, stream>>>(aq, qbuf);
    ln_kernel<<<B_ * CH_, 256, 0, stream>>>(qbuf, snw, snb, nullptr, smallb);
    launch_gemm(1, smallb, 1024, 0, 0, saqT, 1024, 0, 0,
                qkvf, 3072, 0, 0, qkvb, saqb, nullptr, 0,
                B_ * CH_, 3072, 1024, 1, 1, stream);
    launch_gemm(0, qkvb, 3072, (long)CH_ * 3072, 64,
                qkvb + 1024, 3072, (long)CH_ * 3072, 64,
                scoref, CH_, (long)H_ * CH_ * CH_, (long)CH_ * CH_, nullptr,
                nullptr, nullptr, 0, CH_, CH_, HD_, B_ * H_, H_, stream);
    softmax_kernel<<<B_ * H_ * CH_, 256, 0, stream>>>(scoref, Pb, CH_, CH_, 0.125f, 0);
    launch_tc(qkvf + 2048, 3072, (long)CH_ * 3072, 64, H_,
              Vt, CH_, (long)HD_ * CH_, CH_, HD_, B_ * H_, stream);
    launch_gemm(0, Pb, CH_, (long)H_ * CH_ * CH_, (long)CH_ * CH_,
                Vt, CH_, (long)H_ * HD_ * CH_, (long)HD_ * CH_,
                nullptr, 1024, (long)CH_ * D_, 64, attnb,
                nullptr, nullptr, 0, CH_, HD_, CH_, B_ * H_, H_, stream);
    launch_gemm(3, attnb, 1024, 0, 0, saoT, 1024, 0, 0,
                qbuf, 1024, 0, 0, nullptr, saob, qbuf, 1024,
                B_ * CH_, 1024, 1024, 1, 1, stream);

    // ---- cross-attention over comb = [q ; backbone] ----
    copy_rows_kernel<<<B_ * CH_, 256, 0, stream>>>(qbuf, CH_, comb, SC_, 0);
    copy_rows_kernel<<<B_ * S_, 256, 0, stream>>>(bbf, S_, comb, SC_, CH_);
    ln_kernel<<<B_ * SC_, 256, 0, stream>>>(comb, cnw, cnb, nullptr, hb);
    launch_gemm(1, hb, 1024, 0, 0, caqT, 1024, 0, 0,
                qkvf, 3072, 0, 0, qkvb, caqb, nullptr, 0,
                B_ * SC_, 3072, 1024, 1, 1, stream);
    launch_gemm(0, qkvb, 3072, (long)SC_ * 3072, 64,
                qkvb + 1024, 3072, (long)SC_ * 3072, 64,
                scoref, SC_, (long)H_ * SC_ * SC_, (long)SC_ * SC_, nullptr,
                nullptr, nullptr, 0, SC_, SC_, HD_, B_ * H_, H_, stream);
    softmax_kernel<<<B_ * H_ * SC_, 256, 0, stream>>>(scoref, Pb, SC_, SC_, 0.125f, 0);
    launch_tc(qkvf + 2048, 3072, (long)SC_ * 3072, 64, H_,
              Vt, SC_, (long)HD_ * SC_, SC_, HD_, B_ * H_, stream);
    launch_gemm(0, Pb, SC_, (long)H_ * SC_ * SC_, (long)SC_ * SC_,
                Vt, SC_, (long)H_ * HD_ * SC_, (long)HD_ * SC_,
                nullptr, 1024, (long)SC_ * D_, 64, attnb,
                nullptr, nullptr, 0, SC_, HD_, SC_, B_ * H_, H_, stream);
    launch_gemm(1, attnb, 1024, 0, 0, caoT, 1024, 0, 0,
                crossf, 1024, 0, 0, nullptr, caob, nullptr, 0,
                B_ * SC_, 1024, 1024, 1, 1, stream);
    slice_add_kernel<<<B_ * CH_, 256, 0, stream>>>(qbuf, crossf);

    // ---- action FFN + projection ----
    ln_kernel<<<B_ * CH_, 256, 0, stream>>>(qbuf, fnw, fnb, nullptr, smallb);
    launch_gemm(2, smallb, 1024, 0, 0, af1T, 1024, 0, 0,
                nullptr, 4096, 0, 0, ffb, a1b, nullptr, 0,
                B_ * CH_, 4096, 1024, 1, 1, stream);
    launch_gemm(3, ffb, 4096, 0, 0, af2T, 4096, 0, 0,
                qbuf, 1024, 0, 0, nullptr, a2b, qbuf, 1024,
                B_ * CH_, 1024, 4096, 1, 1, stream);
    long nq = (long)B_ * CH_ * D_;
    cvt_kernel<<<(int)((nq + 255) / 256), 256, 0, stream>>>(qbuf, smallb, nq);
    launch_gemm(1, smallb, 1024, 0, 0, apT, 1024, 0, 0,
                (float*)d_out, ADIM_, 0, 0, nullptr, apb, nullptr, 0,
                B_ * CH_, ADIM_, 1024, 1, 1, stream);
}
